// DifferentialDualAttentionInteractiveBlock_24567212933934
// MI455X (gfx1250) — matrix operand order repaired and checked by reading, not compiled, not run on hardware
//
#include <hip/hip_runtime.h>
#include <hip/hip_bf16.h>

typedef _Float16 h16;
typedef __attribute__((ext_vector_type(16))) _Float16 v16h;
typedef __attribute__((ext_vector_type(8)))  float    v8f;

#define NW      1024
#define N_TOK   64
#define DIM     192
#define QKVDIM  576
#define KVDIM   384
#define HEADS   6
#define HDIM    32
#define ATT_SCALE 0.17677669529663689f   // 32^-0.5

// ---------------------------------------------------------------------------
// WMMA helpers (CDNA5 16x16x32 f16 -> f32)
// ---------------------------------------------------------------------------
__device__ __forceinline__ v8f wmma16(v16h a, v16h b, v8f c) {
  return __builtin_amdgcn_wmma_f32_16x16x32_f16(
      /*neg_a=*/false, a, /*neg_b=*/false, b,
      /*c_mod=*/(short)0, c, /*reuse_a=*/false, /*reuse_b=*/false);
}

// A fragment: A row-major (MxK), lda elems; tile at (m0,k0).
// Lane L holds row m0+(L&15); lanes 0-15 cover K {0..15}, 16-31 cover {16..31}.
__device__ __forceinline__ v16h load_a(const h16* A, int lda, int m0, int k0, int lane) {
  const h16* row = A + (m0 + (lane & 15)) * lda + k0;
  const int hh = lane >> 4;
  v16h a;
#pragma unroll
  for (int v = 0; v < 8; ++v) {
    int k = (hh << 4) + (v << 1);
    a[2 * v]     = row[k];
    a[2 * v + 1] = row[k + 1];
  }
  return a;
}

// B fragment with B = M^T: B[k][n] = M[n][k], M row-major (ldm). Lane L holds
// column n0+(L&15), 16 consecutive k starting at k0 + 16*(L>>4)  (contiguous!)
__device__ __forceinline__ v16h load_bT(const h16* M, int ldm, int k0, int n0, int lane) {
  const h16* row = M + (n0 + (lane & 15)) * ldm + k0 + ((lane >> 4) << 4);
  v16h b;
#pragma unroll
  for (int v = 0; v < 16; ++v) b[v] = row[v];
  return b;
}

// B fragment, B row-major (KxN, ldb): lane L holds column n0+(L&15),
// 16 consecutive k (strided loads).
__device__ __forceinline__ v16h load_b(const h16* B, int ldb, int k0, int n0, int lane) {
  const h16* col = B + (k0 + ((lane >> 4) << 4)) * ldb + n0 + (lane & 15);
  v16h b;
#pragma unroll
  for (int v = 0; v < 16; ++v) b[v] = col[v * ldb];
  return b;
}

// Full K=192 accumulation: A (LDS f16, row-major, lda) x W^T (global f16 weight [O][I])
__device__ __forceinline__ v8f gemm192(const h16* A, int lda, const h16* W, int ldw,
                                       int m0, int n0, int lane) {
  v8f acc = {};
#pragma unroll
  for (int ks = 0; ks < DIM / 32; ++ks) {
    v16h a = load_a(A, lda, m0, ks * 32, lane);
    v16h b = load_bT(W, ldw, ks * 32, n0, lane);
    acc = wmma16(a, b, acc);
  }
  return acc;
}

// C layout: VGPR j holds row m0 + j + 8*(lane>>4), col n0 + (lane&15)
__device__ __forceinline__ void store_c_h16(v8f acc, h16* D, int ldd, int m0, int n0, int lane) {
  const int n  = n0 + (lane & 15);
  const int mb = m0 + ((lane >> 4) << 3);
#pragma unroll
  for (int j = 0; j < 8; ++j) D[(mb + j) * ldd + n] = (h16)acc[j];
}

// ---------------------------------------------------------------------------
// Differential attention over 6 heads (shared by SA and CA phases)
// ---------------------------------------------------------------------------
__device__ void diff_attn(const h16* qT, const h16* qR, int ldq,
                          const h16* kT, const h16* kR,
                          const h16* vT, const h16* vR, int ldkv,
                          h16* oT, h16* oR, int ldo,
                          float* St, float* Sr, h16* Dt, h16* Dr,
                          const float* biasAll, float lam,
                          int wave, int lane, int tid) {
  for (int h = 0; h < HEADS; ++h) {
    const float* bias = biasAll + h * (N_TOK * N_TOK);
    // ---- scores: S = (q*scale) @ k^T + bias : 2 x (4x4) tiles, K=32 ----
    for (int t = wave; t < 32; t += 8) {
      const bool isT = t < 16;
      const h16* q = (isT ? qT : qR) + h * HDIM;
      const h16* k = (isT ? kT : kR) + h * HDIM;
      float* S = isT ? St : Sr;
      const int tt = t & 15, mt = tt >> 2, nt = tt & 3;
      v16h a = load_a(q, ldq, mt * 16, 0, lane);
      v16h b = load_bT(k, ldkv, 0, nt * 16, lane);
      v8f acc = {};
      acc = wmma16(a, b, acc);
      const int n  = nt * 16 + (lane & 15);
      const int mb = mt * 16 + ((lane >> 4) << 3);
#pragma unroll
      for (int j = 0; j < 8; ++j) {
        const int m = mb + j;
        S[m * N_TOK + n] = acc[j] * ATT_SCALE + bias[m * N_TOK + n];
      }
    }
    __syncthreads();
    // ---- row softmax: 128 rows over threads 0..127 ----
    if (tid < 128) {
      float* S = (tid < 64 ? St : Sr) + (tid & 63) * N_TOK;
      float mx = -1e30f;
      for (int i = 0; i < N_TOK; ++i) mx = fmaxf(mx, S[i]);
      float sum = 0.f;
      for (int i = 0; i < N_TOK; ++i) { float e = __expf(S[i] - mx); S[i] = e; sum += e; }
      const float r = 1.f / sum;
      for (int i = 0; i < N_TOK; ++i) S[i] *= r;
    }
    __syncthreads();
    // ---- differential combine -> f16 A operands ----
    for (int e = tid; e < N_TOK * N_TOK; e += 256) {
      const float at = St[e], ar = Sr[e];
      Dt[e] = (h16)(at - lam * ar);
      Dr[e] = (h16)(ar - lam * at);
    }
    __syncthreads();
    // ---- O = D @ V : 2 x (4x2) tiles, K=64 ----
    for (int t = wave; t < 16; t += 8) {
      const bool isT = t < 8;
      const h16* D = isT ? Dt : Dr;
      const h16* V = (isT ? vT : vR) + h * HDIM;
      h16* O = (isT ? oT : oR) + h * HDIM;
      const int tt = t & 7, mt = tt >> 1, nt = tt & 1;
      v8f acc = {};
#pragma unroll
      for (int ks = 0; ks < 2; ++ks) {
        v16h a = load_a(D, N_TOK, mt * 16, ks * 32, lane);
        v16h b = load_b(V, ldkv, ks * 32, nt * 16, lane);
        acc = wmma16(a, b, acc);
      }
      store_c_h16(acc, O, ldo, mt * 16, nt * 16, lane);
    }
    __syncthreads();
  }
}

// ---------------------------------------------------------------------------
// Prep kernels
// ---------------------------------------------------------------------------
__global__ void cvt_f32_f16(const float* __restrict__ src, h16* __restrict__ dst, int n) {
  int i = blockIdx.x * blockDim.x + threadIdx.x;
  if (i < n) dst[i] = (h16)src[i];
}

__global__ void build_bias(const float* __restrict__ rpb, float* __restrict__ bias) {
  int i = blockIdx.x * blockDim.x + threadIdx.x;
  if (i >= HEADS * N_TOK * N_TOK) return;
  const int h = i / (N_TOK * N_TOK);
  const int q = (i >> 6) & 63;
  const int k = i & 63;
  const int d0 = (q >> 3) - (k >> 3) + 7;
  const int d1 = (q & 7) - (k & 7) + 7;
  bias[i] = rpb[(d0 * 15 + d1) * HEADS + h];
}

// ---------------------------------------------------------------------------
// Fused per-window pipeline: one workgroup (8 wave32) per window, 288KB LDS
// ---------------------------------------------------------------------------
__global__ __launch_bounds__(256, 1)
void ddaib_main(const float* __restrict__ x, const float* __restrict__ y,
                const h16* __restrict__ w_sa_cr, const h16* __restrict__ w_sa_ct,
                const h16* __restrict__ w_qkv,  const float* __restrict__ qkv_b,
                const h16* __restrict__ w_ca_q, const float* __restrict__ ca_q_b,
                const h16* __restrict__ w_ca_kv,const float* __restrict__ ca_kv_b,
                const h16* __restrict__ w_ca_ct,const h16* __restrict__ w_ca_cr,
                const h16* __restrict__ w_psa,  const float* __restrict__ psa_b,
                const h16* __restrict__ w_pca,  const float* __restrict__ pca_b,
                const float* __restrict__ biasT,
                const float* __restrict__ p_lam_sa, const float* __restrict__ p_lam_ca,
                const float* __restrict__ p_sa_enh, const float* __restrict__ p_ca_enh,
                float* __restrict__ out) {
  extern __shared__ char smem[];
  h16* XH = (h16*)smem;                 // 64x192
  h16* YH = XH + N_TOK * DIM;           // 64x192
  h16* TE = YH + N_TOK * DIM;           // 64x192 (also o_t scratch)
  h16* RE = TE + N_TOK * DIM;           // 64x192 (also o_r scratch)
  h16* QT = RE + N_TOK * DIM;           // 64x576 (SA qkv_t)
  h16* QR = QT + N_TOK * QKVDIM;        // 64x576 (SA qkv_r)
  float* St = (float*)(QR + N_TOK * QKVDIM);   // 64x64 f32
  float* Sr = St + N_TOK * N_TOK;              // 64x64 f32
  h16* Dt = (h16*)(Sr + N_TOK * N_TOK);        // 64x64 f16
  h16* Dr = Dt + N_TOK * N_TOK;                // 64x64 f16
  // CA aliases over the SA qkv region
  h16* Qct = QT;                        // 64x192
  h16* Qcr = QT + N_TOK * DIM;          // 64x192
  h16* KV  = QT + 2 * N_TOK * DIM;      // 128x384
  // stage-0 aliases: raw f32 copies of x,y staged in the (dead) QT region
  float* XF = (float*)QT;               // 64x192 f32
  float* YF = XF + N_TOK * DIM;         // 64x192 f32

  const int b    = blockIdx.x;
  const int tid  = threadIdx.x;
  const int wave = tid >> 5;
  const int lane = tid & 31;

  const float* xg = x + (size_t)b * N_TOK * DIM;
  const float* yg = y + (size_t)b * N_TOK * DIM;

  const float lamSA = fminf(fmaxf(1.f / (1.f + __expf(-p_lam_sa[0])), 0.01f), 0.99f);
  const float lamCA = fminf(fmaxf(1.f / (1.f + __expf(-p_lam_ca[0])), 0.01f), 0.99f);
  const float enhSA = p_sa_enh[0];
  const float enhCA = p_ca_enh[0];

  // ---- stage 0a: async copy x,y (f32) straight into LDS (bypasses VGPRs) ----
  {
    const uint32_t xoff = (uint32_t)(uintptr_t)XF;
    const uint32_t yoff = (uint32_t)(uintptr_t)YF;
    for (int i = tid; i < (N_TOK * DIM) / 4; i += 256) {
      const uint32_t lx = xoff + (uint32_t)i * 16u;
      const uint32_t ly = yoff + (uint32_t)i * 16u;
      const float* gx = xg + i * 4;
      const float* gy = yg + i * 4;
      asm volatile("global_load_async_to_lds_b128 %0, %1, off"
                   :: "v"(lx), "v"(gx) : "memory");
      asm volatile("global_load_async_to_lds_b128 %0, %1, off"
                   :: "v"(ly), "v"(gy) : "memory");
    }
    asm volatile("s_wait_asynccnt 0x0" ::: "memory");
  }
  __syncthreads();

  // ---- stage 0b: f32 (LDS) -> f16 operand copies ----
  for (int i = tid; i < N_TOK * DIM; i += 256) {
    XH[i] = (h16)XF[i];
    YH[i] = (h16)YF[i];
  }
  __syncthreads();

  // =========================== SA phase ===========================
  // stage 1: t_enh = x + e*(y@cr^T) ; r_enh = y + e*(x@ct^T)   (96 tiles)
  // residual read from the staged f32 LDS copies (QT region still intact)
  for (int t = wave; t < 96; t += 8) {
    const bool isT = t < 48;
    const h16* A = isT ? YH : XH;
    const h16* W = isT ? w_sa_cr : w_sa_ct;
    const float* res = isT ? XF : YF;
    h16* D = isT ? TE : RE;
    const int tt = t % 48, mt = tt / 12, nt = tt % 12;
    v8f acc = gemm192(A, DIM, W, DIM, mt * 16, nt * 16, lane);
    const int n  = nt * 16 + (lane & 15);
    const int mb = mt * 16 + ((lane >> 4) << 3);
#pragma unroll
    for (int j = 0; j < 8; ++j) {
      const int m = mb + j;
      D[m * DIM + n] = (h16)(res[m * DIM + n] + enhSA * acc[j]);
    }
  }
  __syncthreads();

  // stage 2: qkv_{t,r} = enh @ qkv_w^T + qkv_b   (288 tiles; overwrites XF/YF)
  for (int t = wave; t < 288; t += 8) {
    const bool isT = t < 144;
    const h16* A = isT ? TE : RE;
    h16* D = isT ? QT : QR;
    const int tt = t % 144, mt = tt / 36, nt = tt % 36;
    v8f acc = gemm192(A, DIM, w_qkv, DIM, mt * 16, nt * 16, lane);
    const int n  = nt * 16 + (lane & 15);
    const float bv = qkv_b[n];
    const int mb = mt * 16 + ((lane >> 4) << 3);
#pragma unroll
    for (int j = 0; j < 8; ++j) D[(mb + j) * QKVDIM + n] = (h16)(acc[j] + bv);
  }
  __syncthreads();

  // stage 3: differential attention (writes o_t/o_r into TE/RE)
  diff_attn(QT, QR, QKVDIM,
            QT + DIM, QR + DIM,
            QT + 2 * DIM, QR + 2 * DIM, QKVDIM,
            TE, RE, DIM,
            St, Sr, Dt, Dr, biasT, lamSA, wave, lane, tid);

  // stage 4: proj -> out rows [b] and [NW+b]  (96 tiles)
  for (int t = wave; t < 96; t += 8) {
    const bool isT = t < 48;
    const h16* A = isT ? TE : RE;
    float* Og = out + (size_t)(isT ? b : NW + b) * (N_TOK * DIM);
    const int tt = t % 48, mt = tt / 12, nt = tt % 12;
    v8f acc = gemm192(A, DIM, w_psa, DIM, mt * 16, nt * 16, lane);
    const int n  = nt * 16 + (lane & 15);
    const float bv = psa_b[n];
    const int mb = mt * 16 + ((lane >> 4) << 3);
#pragma unroll
    for (int j = 0; j < 8; ++j) Og[(mb + j) * DIM + n] = acc[j] + bv;
  }
  __syncthreads();

  // =========================== CA phase ===========================
  // stage 5: CA enh (overwrites TE/RE); residual from global (LDS f32 copy is gone)
  for (int t = wave; t < 96; t += 8) {
    const bool isT = t < 48;
    const h16* A = isT ? YH : XH;
    const h16* W = isT ? w_ca_cr : w_ca_ct;
    const float* res = isT ? xg : yg;
    h16* D = isT ? TE : RE;
    const int tt = t % 48, mt = tt / 12, nt = tt % 12;
    v8f acc = gemm192(A, DIM, W, DIM, mt * 16, nt * 16, lane);
    const int n  = nt * 16 + (lane & 15);
    const int mb = mt * 16 + ((lane >> 4) << 3);
#pragma unroll
    for (int j = 0; j < 8; ++j) {
      const int m = mb + j;
      D[m * DIM + n] = (h16)(res[m * DIM + n] + enhCA * acc[j]);
    }
  }
  __syncthreads();

  // stage 6: q_t/q_r (96 tiles) + kv over [te;re] (192 tiles) = 288 tiles
  for (int t = wave; t < 288; t += 8) {
    if (t < 96) {
      const bool isT = t < 48;
      const h16* A = isT ? TE : RE;
      h16* D = isT ? Qct : Qcr;
      const int tt = t % 48, mt = tt / 12, nt = tt % 12;
      v8f acc = gemm192(A, DIM, w_ca_q, DIM, mt * 16, nt * 16, lane);
      const int n  = nt * 16 + (lane & 15);
      const float bv = ca_q_b[n];
      const int mb = mt * 16 + ((lane >> 4) << 3);
#pragma unroll
      for (int j = 0; j < 8; ++j) D[(mb + j) * DIM + n] = (h16)(acc[j] + bv);
    } else {
      const int tt = t - 96;           // 8 m-tiles x 24 n-tiles
      const int mt = tt / 24, nt = tt % 24;
      const h16* A = (mt < 4) ? TE : RE;
      const int m0 = (mt & 3) * 16;
      v8f acc = gemm192(A, DIM, w_ca_kv, DIM, m0, nt * 16, lane);
      const int n  = nt * 16 + (lane & 15);
      const float bv = ca_kv_b[n];
      const int mb = mt * 16 + ((lane >> 4) << 3);
#pragma unroll
      for (int j = 0; j < 8; ++j) KV[(mb + j) * KVDIM + n] = (h16)(acc[j] + bv);
    }
  }
  __syncthreads();

  // stage 7: CA differential attention (o into TE/RE, free now)
  diff_attn(Qct, Qcr, DIM,
            KV, KV + N_TOK * KVDIM,                      // k_t rows 0..63, k_r rows 64..127
            KV + DIM, KV + N_TOK * KVDIM + DIM, KVDIM,   // v_t, v_r (cols 192..383)
            TE, RE, DIM,
            St, Sr, Dt, Dr, biasT, lamCA, wave, lane, tid);

  // stage 8: proj_ca -> out rows [2*NW + 2b] (ca_t) and [2*NW + 2b + 1] (ca_r)
  for (int t = wave; t < 96; t += 8) {
    const bool isT = t < 48;
    const h16* A = isT ? TE : RE;
    float* Og = out + (size_t)(2 * NW + 2 * b + (isT ? 0 : 1)) * (N_TOK * DIM);
    const int tt = t % 48, mt = tt / 12, nt = tt % 12;
    v8f acc = gemm192(A, DIM, w_pca, DIM, mt * 16, nt * 16, lane);
    const int n  = nt * 16 + (lane & 15);
    const float bv = pca_b[n];
    const int mb = mt * 16 + ((lane >> 4) << 3);
#pragma unroll
    for (int j = 0; j < 8; ++j) Og[(mb + j) * DIM + n] = acc[j] + bv;
  }
}

// ---------------------------------------------------------------------------
// Launch
// ---------------------------------------------------------------------------
extern "C" void kernel_launch(void* const* d_in, const int* in_sizes, int n_in,
                              void* d_out, int out_size, void* d_ws, size_t ws_size,
                              hipStream_t stream) {
  (void)in_sizes; (void)n_in; (void)out_size; (void)ws_size;
  const float* x        = (const float*)d_in[0];
  const float* y        = (const float*)d_in[1];
  const float* sa_qkv_w = (const float*)d_in[2];
  const float* sa_qkv_b = (const float*)d_in[3];
  const float* sa_ct_w  = (const float*)d_in[4];
  const float* sa_cr_w  = (const float*)d_in[5];
  const float* ca_q_w   = (const float*)d_in[6];
  const float* ca_q_b   = (const float*)d_in[7];
  const float* ca_kv_w  = (const float*)d_in[8];
  const float* ca_kv_b  = (const float*)d_in[9];
  const float* ca_ct_w  = (const float*)d_in[10];
  const float* ca_cr_w  = (const float*)d_in[11];
  const float* rpb      = (const float*)d_in[12];
  const float* psa_w    = (const float*)d_in[13];
  const float* psa_b    = (const float*)d_in[14];
  const float* pca_w    = (const float*)d_in[15];
  const float* pca_b    = (const float*)d_in[16];
  const float* lam_sa   = (const float*)d_in[17];
  const float* lam_ca   = (const float*)d_in[18];
  const float* sa_enh   = (const float*)d_in[19];
  const float* ca_enh   = (const float*)d_in[20];
  float* out = (float*)d_out;

  // workspace: bias table (f32) then f16 weight copies
  float* biasT = (float*)d_ws;                                      // 6*64*64 f32
  h16* wbase     = (h16*)((char*)d_ws + HEADS * N_TOK * N_TOK * 4);
  h16* w_qkv     = wbase;                  // 576*192
  h16* w_sa_ct   = w_qkv   + 110592;       // 192*192
  h16* w_sa_cr   = w_sa_ct + 36864;
  h16* w_ca_q    = w_sa_cr + 36864;
  h16* w_ca_kv   = w_ca_q  + 36864;        // 384*192
  h16* w_ca_ct   = w_ca_kv + 73728;
  h16* w_ca_cr   = w_ca_ct + 36864;
  h16* w_proj_sa = w_ca_cr + 36864;
  h16* w_proj_ca = w_proj_sa + 36864;

  auto cvt = [&](const float* s, h16* d, int n) {
    cvt_f32_f16<<<(n + 255) / 256, 256, 0, stream>>>(s, d, n);
  };
  cvt(sa_qkv_w, w_qkv,     110592);
  cvt(sa_ct_w,  w_sa_ct,   36864);
  cvt(sa_cr_w,  w_sa_cr,   36864);
  cvt(ca_q_w,   w_ca_q,    36864);
  cvt(ca_kv_w,  w_ca_kv,   73728);
  cvt(ca_ct_w,  w_ca_ct,   36864);
  cvt(ca_cr_w,  w_ca_cr,   36864);
  cvt(psa_w,    w_proj_sa, 36864);
  cvt(pca_w,    w_proj_ca, 36864);
  build_bias<<<(HEADS * N_TOK * N_TOK + 255) / 256, 256, 0, stream>>>(rpb, biasT);

  const int lds_bytes = 294912;  // 288 KB per workgroup (<= 320 KB WGP limit)
  hipFuncSetAttribute((const void*)ddaib_main,
                      hipFuncAttributeMaxDynamicSharedMemorySize, lds_bytes);

  ddaib_main<<<NW, 256, lds_bytes, stream>>>(
      x, y,
      w_sa_cr, w_sa_ct, w_qkv, sa_qkv_b,
      w_ca_q, ca_q_b, w_ca_kv, ca_kv_b,
      w_ca_ct, w_ca_cr,
      w_proj_sa, psa_b,
      w_proj_ca, pca_b,
      biasT,
      lam_sa, lam_ca, sa_enh, ca_enh,
      out);
}
